// SPEmbedder2Conv_21062519620293
// MI455X (gfx1250) — matrix operand order repaired and checked by reading, not compiled, not run on hardware
//
#include <hip/hip_runtime.h>
#include <hip/hip_bf16.h>

// ---------------- problem constants (from reference) ----------------
#define NG      100           // graphs
#define NPG     1000          // nodes per graph
#define NN      (NG * NPG)    // 100000 nodes
#define DEG     16
#define NE      (NN * DEG)    // 1600000 edges
#define INF     64            // IN_FEATS
#define HID     128           // HIDDEN
#define RD      64            // READOUT_DIM
#define SLOPE   0.01f
#define EPSN    1e-5f
#define INVCNT  (1.0f / NPG)
#define OUTW    384           // 64 + 128 + 64 + 128

typedef __attribute__((ext_vector_type(16))) __bf16   v16bf;
typedef __attribute__((ext_vector_type(8)))  float    v8f;
typedef __attribute__((ext_vector_type(8)))  unsigned v8u;

__device__ __forceinline__ float leaky(float x) { return x >= 0.f ? x : SLOPE * x; }

__device__ __forceinline__ unsigned bfbits(float x) {
    union { float f; unsigned u; } v; v.f = x;
    return (v.u + 0x7FFFu + ((v.u >> 16) & 1u)) >> 16;   // RNE
}
__device__ __forceinline__ unsigned pack2bf(float lo, float hi) {
    return bfbits(lo) | (bfbits(hi) << 16);
}

// ---------------- degree / rsqrt ----------------
__global__ void deg_kernel(const int* __restrict__ src, const int* __restrict__ dst,
                           float* __restrict__ odeg, float* __restrict__ ideg) {
    int e = blockIdx.x * blockDim.x + threadIdx.x;
    if (e >= NE) return;
    atomicAdd(&odeg[src[e]], 1.0f);
    atomicAdd(&ideg[dst[e]], 1.0f);
}

__global__ void isqrt_kernel(float* __restrict__ p, int n) {
    int i = blockIdx.x * blockDim.x + threadIdx.x;
    if (i < n) p[i] = rsqrtf(fmaxf(p[i], 1.0f));
}

// ---------------- bf16 packing ----------------
// Activations: row-major f32 [NN,KIN] -> row-major packed bf16 pairs (u32),
// with optional per-row scale fused in. kshift = log2(KIN/2).
__global__ void packA_kernel(const float* __restrict__ X, const float* __restrict__ rowscale,
                             unsigned* __restrict__ Xp, int kshift) {
    size_t idx = (size_t)blockIdx.x * blockDim.x + threadIdx.x;   // NN*KIN/2 threads
    if (idx >= ((size_t)NN << kshift)) return;
    int row = (int)(idx >> kshift);
    const float2 v = *(const float2*)(X + 2 * idx);
    float s = rowscale ? rowscale[row] : 1.0f;
    Xp[idx] = pack2bf(v.x * s, v.y * s);
}

// Weights: f32 [KIN,128] -> fragment-ordered bf16 dwords:
// Wp[kchunk][wave(8)][lane(32)][j(8)], dword j of the B fragment holds
// K = kchunk*32 + 2*(j&3) + 8*h + 16*(j>>2) (and +1), N = wave*16 + (lane&15).
__global__ void packW_kernel(const float* __restrict__ W, unsigned* __restrict__ Wp, int KIN) {
    int idx = blockIdx.x * blockDim.x + threadIdx.x;   // (KIN/32)*2048 threads
    if (idx >= (KIN >> 5) * 2048) return;
    int j     = idx & 7;
    int lane  = (idx >> 3) & 31;
    int wave  = (idx >> 8) & 7;
    int kch   = idx >> 11;
    int h = lane >> 4, l = lane & 15;
    int kb  = kch * 32 + 2 * (j & 3) + 8 * h + 16 * (j >> 2);
    int col = wave * 16 + l;
    Wp[idx] = pack2bf(W[(size_t)kb * HID + col], W[(size_t)(kb + 1) * HID + col]);
}

// ---------------- WMMA GEMM: Y[NN,128] = Xbf[NN,KIN] @ Wbf[KIN,128] ----------------
// One wave per 16 output cols; two 16-row tiles per wave (B fragment reused).
__global__ __launch_bounds__(256)
void wmma_gemm(const unsigned* __restrict__ Xp, const unsigned* __restrict__ Wp,
               float* __restrict__ Y, int KIN) {
    const int wave = threadIdx.x >> 5;       // 0..7 -> column tile
    const int lane = threadIdx.x & 31;
    const int h = lane >> 4, l = lane & 15;
    const int rowbase = blockIdx.x * 32;     // two 16-row tiles
    const int colbase = wave * 16;
    const int KH = KIN >> 1;                 // dwords per packed row

    v8f c0 = {}, c1 = {};
    for (int kk = 0; kk < KIN; kk += 32) {
        const int kch = kk >> 5;
        // B fragment: 8 contiguous dwords per lane
        const uint4* bp = (const uint4*)(Wp + (((size_t)(kch * 8 + wave) * 32 + lane) << 3));
        uint4 bq0 = bp[0], bq1 = bp[1];
        v16bf b = __builtin_bit_cast(v16bf,
                    (v8u){bq0.x, bq0.y, bq0.z, bq0.w, bq1.x, bq1.y, bq1.z, bq1.w});
        // A fragment, tile 0: dwords [kk/2+4h .. +3] and [kk/2+8+4h .. +3]
        const unsigned* ar0 = Xp + (size_t)(rowbase + l) * KH + (kk >> 1) + 4 * h;
        uint4 a0 = *(const uint4*)ar0;
        uint4 a1 = *(const uint4*)(ar0 + 8);
        v16bf a = __builtin_bit_cast(v16bf,
                    (v8u){a0.x, a0.y, a0.z, a0.w, a1.x, a1.y, a1.z, a1.w});
        c0 = __builtin_amdgcn_wmma_f32_16x16x32_bf16(false, a, false, b,
                                                     (short)0, c0, false, false);
        // A fragment, tile 1 (rows +16)
        const unsigned* ar1 = Xp + (size_t)(rowbase + 16 + l) * KH + (kk >> 1) + 4 * h;
        uint4 a2 = *(const uint4*)ar1;
        uint4 a3 = *(const uint4*)(ar1 + 8);
        v16bf a_ = __builtin_bit_cast(v16bf,
                    (v8u){a2.x, a2.y, a2.z, a2.w, a3.x, a3.y, a3.z, a3.w});
        c1 = __builtin_amdgcn_wmma_f32_16x16x32_bf16(false, a_, false, b,
                                                     (short)0, c1, false, false);
    }
#pragma unroll
    for (int r = 0; r < 8; ++r) {            // C/D: VGPR r -> M = r + 8*h, N = lane&15
        Y[(size_t)(rowbase + 8 * h + r) * HID + colbase + l]      = c0[r];
        Y[(size_t)(rowbase + 16 + 8 * h + r) * HID + colbase + l] = c1[r];
    }
}

// ---------------- edge gather * ew -> scatter-add ----------------
__global__ void scatter_kernel(const float* __restrict__ H, const float* __restrict__ ew,
                               const int* __restrict__ src, const int* __restrict__ dst,
                               float* __restrict__ agg) {
    size_t t = (size_t)blockIdx.x * blockDim.x + threadIdx.x;  // NE*32 threads
    size_t e = t >> 5;
    int f = (int)(t & 31) * 4;
    if (e >= NE) return;
    int si = src[e], di = dst[e];
    float w = ew[e];
    const float4 v = *(const float4*)(H + (size_t)si * HID + f);
    float* ap = agg + (size_t)di * HID + f;
    atomicAdd(ap + 0, v.x * w);
    atomicAdd(ap + 1, v.y * w);
    atomicAdd(ap + 2, v.z * w);
    atomicAdd(ap + 3, v.w * w);
}

// ---------------- elementwise helpers over [NN,128] ----------------
__global__ void rowscale_kernel(float* __restrict__ x, const float* __restrict__ s) {
    size_t idx = (size_t)blockIdx.x * blockDim.x + threadIdx.x;
    if (idx >= (size_t)NN * HID) return;
    x[idx] *= s[idx >> 7];
}

__global__ void mean_accum_kernel(const float* __restrict__ x, float* __restrict__ gm) {
    size_t idx = (size_t)blockIdx.x * blockDim.x + threadIdx.x;
    if (idx >= (size_t)NN * HID) return;
    int i = (int)(idx >> 7), f = (int)(idx & 127);
    atomicAdd(&gm[(i / NPG) * HID + f], x[idx]);
}

__global__ void scale_small_kernel(float* __restrict__ p, float c, int n) {
    int i = blockIdx.x * blockDim.x + threadIdx.x;
    if (i < n) p[i] *= c;
}

__global__ void subvar_kernel(float* __restrict__ x, const float* __restrict__ alpha,
                              const float* __restrict__ gm, float* __restrict__ gv) {
    size_t idx = (size_t)blockIdx.x * blockDim.x + threadIdx.x;
    if (idx >= (size_t)NN * HID) return;
    int i = (int)(idx >> 7), f = (int)(idx & 127), g = i / NPG;
    float sub = x[idx] - alpha[f] * gm[g * HID + f];
    x[idx] = sub;
    atomicAdd(&gv[g * HID + f], sub * sub);
}

__global__ void norm_final_kernel(const float* __restrict__ sub, const float* __restrict__ gamma,
                                  const float* __restrict__ beta, const float* __restrict__ gv,
                                  float* __restrict__ dst) {
    size_t idx = (size_t)blockIdx.x * blockDim.x + threadIdx.x;
    if (idx >= (size_t)NN * HID) return;
    int i = (int)(idx >> 7), f = (int)(idx & 127), g = i / NPG;
    float v = gamma[f] * sub[idx] * rsqrtf(gv[g * HID + f] + EPSN) + beta[f];
    dst[idx] = leaky(v);
}

// pooled += leaky(P + b1); msum += h   (means taken later)
__global__ void pool_kernel(const float* __restrict__ P, const float* __restrict__ b1,
                            const float* __restrict__ h, float* __restrict__ pooled,
                            float* __restrict__ msum) {
    size_t idx = (size_t)blockIdx.x * blockDim.x + threadIdx.x;
    if (idx >= (size_t)NN * HID) return;
    int i = (int)(idx >> 7), f = (int)(idx & 127), g = i / NPG;
    atomicAdd(&pooled[g * HID + f], leaky(P[idx] + b1[f]));
    atomicAdd(&msum[g * HID + f], h[idx]);
}

// r = leaky(mean(pooled) @ w2 + b2)  -> out[:, col_off : col_off+RD]
__global__ void head_kernel(const float* __restrict__ pooled, const float* __restrict__ w2,
                            const float* __restrict__ b2, float* __restrict__ out, int col_off) {
    int idx = blockIdx.x * blockDim.x + threadIdx.x;
    if (idx >= NG * RD) return;
    int g = idx / RD, o = idx % RD;
    float acc = b2[o];
    for (int f = 0; f < HID; ++f)
        acc += pooled[g * HID + f] * INVCNT * w2[f * RD + o];
    out[g * OUTW + col_off + o] = leaky(acc);
}

__global__ void mwrite_kernel(const float* __restrict__ msum, float* __restrict__ out, int col_off) {
    int idx = blockIdx.x * blockDim.x + threadIdx.x;
    if (idx >= NG * HID) return;
    int g = idx / HID, f = idx % HID;
    out[g * OUTW + col_off + f] = msum[idx] * INVCNT;
}

__global__ void final_leaky_kernel(float* __restrict__ out) {
    int idx = blockIdx.x * blockDim.x + threadIdx.x;
    if (idx < NG * OUTW) out[idx] = leaky(out[idx]);
}

// ---------------------------------------------------------------------------
static inline void run_gemm(const float* X, const float* W, const float* rowscale,
                            unsigned* Xp, unsigned* Wp, float* Y, int KIN,
                            hipStream_t stream) {
    const int T = 256;
    const int kshift = (KIN == 64) ? 5 : 6;          // log2(KIN/2)
    const int nPack  = (int)(((size_t)NN * KIN / 2) / T);   // exact (12500 / 25000)
    packW_kernel<<<((KIN >> 5) * 2048) / T, T, 0, stream>>>(W, Wp, KIN);
    packA_kernel<<<nPack, T, 0, stream>>>(X, rowscale, Xp, kshift);
    wmma_gemm<<<NN / 32, T, 0, stream>>>(Xp, Wp, Y, KIN);
}

extern "C" void kernel_launch(void* const* d_in, const int* in_sizes, int n_in,
                              void* d_out, int out_size, void* d_ws, size_t ws_size,
                              hipStream_t stream) {
    const float* x     = (const float*)d_in[0];
    const float* ew    = (const float*)d_in[1];
    const float* W1    = (const float*)d_in[2];
    const float* W2    = (const float*)d_in[3];
    const float* gn1a  = (const float*)d_in[4];
    const float* gn1g  = (const float*)d_in[5];
    const float* gn1b  = (const float*)d_in[6];
    const float* gn2a  = (const float*)d_in[7];
    const float* gn2g  = (const float*)d_in[8];
    const float* gn2b  = (const float*)d_in[9];
    const float* r1w1  = (const float*)d_in[10];
    const float* r1b1  = (const float*)d_in[11];
    const float* r1w2  = (const float*)d_in[12];
    const float* r1b2  = (const float*)d_in[13];
    const float* r2w1  = (const float*)d_in[14];
    const float* r2b1  = (const float*)d_in[15];
    const float* r2w2  = (const float*)d_in[16];
    const float* r2b2  = (const float*)d_in[17];
    const int*   src   = (const int*)d_in[18];
    const int*   dst   = (const int*)d_in[19];
    float*       out   = (float*)d_out;

    const size_t NH = (size_t)NN * HID;
    float*    bufA   = (float*)d_ws;
    float*    bufB   = bufA + NH;
    float*    oisq   = bufB + NH;
    float*    iisq   = oisq + NN;
    float*    gmean  = iisq + NN;
    float*    gvar   = gmean + NG * HID;
    float*    pooled = gvar  + NG * HID;
    float*    msum   = pooled + NG * HID;
    unsigned* Xp     = (unsigned*)(msum + NG * HID);       // NN * 64 dwords (bf16 pairs)
    unsigned* Wp     = Xp + NH / 2;                        // up to 8192 dwords

    const int T = 256;
    const int bNH  = (int)(NH / T);                        // 50000, exact
    const int bE   = NE / T;                               // 6250, exact
    const int bE32 = (int)(((size_t)NE * 32) / T);         // 200000, exact
    const int bGH  = (NG * HID + T - 1) / T;
    const int bGR  = (NG * RD  + T - 1) / T;

    // zero accumulators (deg buffers, stats, layer-1 scatter target)
    hipMemsetAsync(oisq,  0, (size_t)2 * NN * sizeof(float), stream);       // oisq+iisq
    hipMemsetAsync(gmean, 0, (size_t)4 * NG * HID * sizeof(float), stream); // gmean..msum
    hipMemsetAsync(bufB,  0, NH * sizeof(float), stream);

    // degrees -> rsqrt(clip(deg,1))
    deg_kernel<<<bE, T, 0, stream>>>(src, dst, oisq, iisq);
    isqrt_kernel<<<(2 * NN + T - 1) / T, T, 0, stream>>>(oisq, 2 * NN);

    // ---- layer 1: conv ----
    run_gemm(x, W1, oisq, Xp, Wp, bufA, INF, stream);
    scatter_kernel<<<bE32, T, 0, stream>>>(bufA, ew, src, dst, bufB);
    rowscale_kernel<<<bNH, T, 0, stream>>>(bufB, iisq);
    // graph norm 1 (result -> bufA = h1)
    mean_accum_kernel<<<bNH, T, 0, stream>>>(bufB, gmean);
    scale_small_kernel<<<bGH, T, 0, stream>>>(gmean, INVCNT, NG * HID);
    subvar_kernel<<<bNH, T, 0, stream>>>(bufB, gn1a, gmean, gvar);
    scale_small_kernel<<<bGH, T, 0, stream>>>(gvar, INVCNT, NG * HID);
    norm_final_kernel<<<bNH, T, 0, stream>>>(bufB, gn1g, gn1b, gvar, bufA);
    // readout 1 + m1
    run_gemm(bufA, r1w1, nullptr, Xp, Wp, bufB, HID, stream);
    pool_kernel<<<bNH, T, 0, stream>>>(bufB, r1b1, bufA, pooled, msum);
    head_kernel<<<bGR, T, 0, stream>>>(pooled, r1w2, r1b2, out, 0);
    mwrite_kernel<<<bGH, T, 0, stream>>>(msum, out, RD);

    // ---- layer 2: conv ----
    run_gemm(bufA, W2, oisq, Xp, Wp, bufB, HID, stream);
    hipMemsetAsync(bufA, 0, NH * sizeof(float), stream);
    hipMemsetAsync(gmean, 0, (size_t)4 * NG * HID * sizeof(float), stream);
    scatter_kernel<<<bE32, T, 0, stream>>>(bufB, ew, src, dst, bufA);
    rowscale_kernel<<<bNH, T, 0, stream>>>(bufA, iisq);
    // graph norm 2 (in place in bufA = h2)
    mean_accum_kernel<<<bNH, T, 0, stream>>>(bufA, gmean);
    scale_small_kernel<<<bGH, T, 0, stream>>>(gmean, INVCNT, NG * HID);
    subvar_kernel<<<bNH, T, 0, stream>>>(bufA, gn2a, gmean, gvar);
    scale_small_kernel<<<bGH, T, 0, stream>>>(gvar, INVCNT, NG * HID);
    norm_final_kernel<<<bNH, T, 0, stream>>>(bufA, gn2g, gn2b, gvar, bufA);
    // readout 2 + m2
    run_gemm(bufA, r2w1, nullptr, Xp, Wp, bufB, HID, stream);
    pool_kernel<<<bNH, T, 0, stream>>>(bufB, r2b1, bufA, pooled, msum);
    head_kernel<<<bGR, T, 0, stream>>>(pooled, r2w2, r2b2, out, RD + HID);
    mwrite_kernel<<<bGH, T, 0, stream>>>(msum, out, RD + HID + RD);

    final_leaky_kernel<<<(NG * OUTW + T - 1) / T, T, 0, stream>>>(out);
}
